// CGFE_36404142801270
// MI455X (gfx1250) — matrix operand order repaired and checked by reading, not compiled, not run on hardware
//
#include <hip/hip_runtime.h>
#include <hip/hip_bf16.h>
#include <stdint.h>

typedef __bf16 bf16_t;
typedef __attribute__((ext_vector_type(16))) __bf16 v16bf;
typedef __attribute__((ext_vector_type(8)))  float  v8f;

#define BATCH 4
#define CCH   256
#define NPIX  4096   // 64*64
#define DQK   32

// LDS layout (dynamic): padded rows to avoid bank conflicts on b128 frag reads
#define KT_ROW   40                     // 32 bf16 + 8 pad  (80 B row)
#define VT_ROW   72                     // 64 bf16 + 8 pad  (144 B row)
#define PL_ROW   68                     // 64 f32 + 4 pad   (272 B row)
#define KT_BYTES (2 * 64 * KT_ROW * 2)          // 10240
#define VT_BYTES (2 * 256 * VT_ROW * 2)         // 73728
#define PL_BYTES (4 * 16 * PL_ROW * 4)          // 17408
#define SMEM_BYTES (KT_BYTES + VT_BYTES + PL_BYTES)

// ---------------------------------------------------------------------------
// Projection: out[o,n] = sum_c W[o,c] * x[b,c,n] + bias[o], cast to bf16.
// transpose_out=1 stores [B][N][32] (q/k, Cout==32); else [B][Cout][N] (v).
// ---------------------------------------------------------------------------
__global__ __launch_bounds__(256) void proj_bf16_kernel(
    const float* __restrict__ x, const float* __restrict__ W,
    const float* __restrict__ bias, bf16_t* __restrict__ out,
    int Cout, int transpose_out)
{
    __shared__ float xt[64][64];   // [cin][pixel]
    __shared__ float wt[32][64];   // [cout][cin]

    const int t     = threadIdx.x;
    const int b     = blockIdx.x >> 6;
    const int n0    = (blockIdx.x & 63) * 64;
    const int obase = blockIdx.y * 32;
    const int px    = t & 63;
    const int og    = t >> 6;

    float acc[8];
#pragma unroll
    for (int oi = 0; oi < 8; ++oi) acc[oi] = 0.f;

    for (int cb = 0; cb < CCH; cb += 64) {
        for (int idx = t; idx < 64 * 64; idx += 256) {
            int cc = idx >> 6, pp = idx & 63;
            xt[cc][pp] = x[((size_t)b * CCH + (cb + cc)) * NPIX + n0 + pp];
        }
        for (int idx = t; idx < 32 * 64; idx += 256) {
            int oo = idx >> 6, cc = idx & 63;
            wt[oo][cc] = W[(size_t)(obase + oo) * CCH + cb + cc];
        }
        __syncthreads();
#pragma unroll 4
        for (int c = 0; c < 64; ++c) {
            float xc = xt[c][px];
#pragma unroll
            for (int oi = 0; oi < 8; ++oi)
                acc[oi] += wt[og * 8 + oi][c] * xc;
        }
        __syncthreads();
    }

#pragma unroll
    for (int oi = 0; oi < 8; ++oi) {
        int o   = obase + og * 8 + oi;
        float r = acc[oi] + bias[o];
        if (transpose_out)
            out[((size_t)b * NPIX + n0 + px) * DQK + o] = (bf16_t)r;
        else
            out[((size_t)b * Cout + o) * NPIX + n0 + px] = (bf16_t)r;
    }
}

// ---------------------------------------------------------------------------
// Async global->LDS helper (gfx1250, 16 B per lane, tracked by ASYNCcnt).
// ---------------------------------------------------------------------------
__device__ __forceinline__ void async_b128_to_lds(unsigned lds_off,
                                                  const void* gaddr) {
    asm volatile("global_load_async_to_lds_b128 %0, %1, off"
                 :: "v"(lds_off), "v"(gaddr) : "memory");
}
__device__ __forceinline__ void wait_async0() {
    asm volatile("s_wait_asynccnt 0x0" ::: "memory");
}

// ---------------------------------------------------------------------------
// Fused flash attention: out = x + gamma * softmax(q^T k) @ v^T  (per batch).
// 256 threads = 8 waves: wave = (row-tile m = w>>1, channel-half h = w&1).
// K/V tiles double-buffered in LDS via async loads.
// ---------------------------------------------------------------------------
__global__ __launch_bounds__(256) void attn_flash_kernel(
    const bf16_t* __restrict__ qT,   // [B][N][32]
    const bf16_t* __restrict__ kT,   // [B][N][32]
    const bf16_t* __restrict__ v1,   // [B][C][N]
    const bf16_t* __restrict__ v2,
    const float*  __restrict__ x1, const float* __restrict__ x2,
    const float*  __restrict__ g1, const float* __restrict__ g2,
    float* __restrict__ out)
{
    extern __shared__ __align__(16) char smem[];
    bf16_t* Kt = (bf16_t*)smem;                       // [2][64][KT_ROW]
    bf16_t* Vt = (bf16_t*)(smem + KT_BYTES);          // [2][256][VT_ROW]
    float*  Pl = (float*)(smem + KT_BYTES + VT_BYTES);// [4][16][PL_ROW]

    const int t    = threadIdx.x;
    const int lane = t & 31;
    const int w    = t >> 5;
    const int m    = w >> 1;
    const int h    = w & 1;
    const int b    = blockIdx.x >> 6;
    const int i0   = (blockIdx.x & 63) * 64 + 16 * m;
    const int vsel = blockIdx.y;

    const bf16_t* vmat = vsel ? v2 : v1;
    const float*  xres = vsel ? x2 : x1;
    const float   gamma = vsel ? g2[0] : g1[0];

    const int ln16  = lane & 15;
    const int lhi   = lane >> 4;
    const int d0    = lhi * 8;
    const int cbase = h * 128;

    const bf16_t* kTb  = kT + (size_t)b * NPIX * DQK;
    const bf16_t* vbat = vmat + (size_t)b * CCH * NPIX;

    // Cooperative async prefetch of one (K,V) j-tile into buffer `buf`.
    const int vc   = t >> 3;        // 0..31 base channel group (8 chunks/row)
    const int vpart= t & 7;         // 16B part within a 64-col row
    const int kj   = t >> 2;        // 0..63 row of K tile
    const int kpart= t & 3;         // 16B part within 32-el K row
    auto prefetch = [&](int buf, int j0n) {
#pragma unroll
        for (int s = 0; s < 8; ++s) {
            const int c = vc + s * 32;
            const bf16_t* g = vbat + (size_t)c * NPIX + j0n + vpart * 8;
            unsigned lo = (unsigned)(size_t)&Vt[((buf * 256) + c) * VT_ROW + vpart * 8];
            async_b128_to_lds(lo, g);
        }
        {
            const bf16_t* g = kTb + (size_t)(j0n + kj) * DQK + kpart * 8;
            unsigned lo = (unsigned)(size_t)&Kt[((buf * 64) + kj) * KT_ROW + kpart * 8];
            async_b128_to_lds(lo, g);
        }
    };

    // --- Q A-fragment, resident for whole kernel ---
    const bf16_t* qb = qT + ((size_t)b * NPIX + (i0 + ln16)) * DQK;
    v16bf qfrag;
    {
        union { v16bf f; uint4 u[2]; } r;
        r.u[0] = *(const uint4*)(qb + d0);
        r.u[1] = *(const uint4*)(qb + 16 + d0);
        qfrag = r.f;
    }

    const v8f zero8 = {0.f, 0.f, 0.f, 0.f, 0.f, 0.f, 0.f, 0.f};
    v8f accO[8];
#pragma unroll
    for (int ct = 0; ct < 8; ++ct) accO[ct] = zero8;
    float mrow[8], lrow[8];
#pragma unroll
    for (int r = 0; r < 8; ++r) { mrow[r] = -1e30f; lrow[r] = 0.f; }

    prefetch(0, 0);
    wait_async0();
    __syncthreads();
    int cur = 0;

    for (int jt = 0; jt < 64; ++jt) {
        const int j0 = jt * 64;
        if (jt < 63) prefetch(cur ^ 1, j0 + 64);   // stream next tile

        // ---- S = Q * K^T : 16 x 64 via 4 WMMAs (K frags from LDS) ----
        v8f S[4];
#pragma unroll
        for (int tt = 0; tt < 4; ++tt) {
            const bf16_t* kb = &Kt[((cur * 64) + tt * 16 + ln16) * KT_ROW + lhi * 8];
            union { v16bf f; uint4 u[2]; } r;
            r.u[0] = *(const uint4*)(kb);
            r.u[1] = *(const uint4*)(kb + 16);
            S[tt] = __builtin_amdgcn_wmma_f32_16x16x32_bf16(
                false, qfrag, false, r.f, (short)0, zero8, false, false);
        }

        // ---- online softmax ----
        float alpha[8];
#pragma unroll
        for (int r = 0; r < 8; ++r) {
            float mx = fmaxf(fmaxf(S[0][r], S[1][r]), fmaxf(S[2][r], S[3][r]));
            mx = fmaxf(mx, __shfl_xor(mx, 1, 32));
            mx = fmaxf(mx, __shfl_xor(mx, 2, 32));
            mx = fmaxf(mx, __shfl_xor(mx, 4, 32));
            mx = fmaxf(mx, __shfl_xor(mx, 8, 32));
            const float mn = fmaxf(mrow[r], mx);
            alpha[r] = __expf(mrow[r] - mn);
            float ps = 0.f;
#pragma unroll
            for (int tt = 0; tt < 4; ++tt) {
                float p = __expf(S[tt][r] - mn);
                S[tt][r] = p;
                ps += p;
            }
            ps += __shfl_xor(ps, 1, 32);
            ps += __shfl_xor(ps, 2, 32);
            ps += __shfl_xor(ps, 4, 32);
            ps += __shfl_xor(ps, 8, 32);
            lrow[r] = lrow[r] * alpha[r] + ps;
            mrow[r] = mn;
        }
#pragma unroll
        for (int ct = 0; ct < 8; ++ct)
#pragma unroll
            for (int r = 0; r < 8; ++r) accO[ct][r] *= alpha[r];

        // ---- transpose P: C-layout -> A-layout via LDS ----
        __syncthreads();
        if (h == 0) {
#pragma unroll
            for (int tt = 0; tt < 4; ++tt)
#pragma unroll
                for (int r = 0; r < 8; ++r)
                    Pl[((m * 16) + r + 8 * lhi) * PL_ROW + tt * 16 + ln16] = S[tt][r];
        }
        __syncthreads();

        // ---- O += P (16x64) * V (64x128-half) : 16 WMMAs (V frags from LDS) ----
#pragma unroll
        for (int kc = 0; kc < 2; ++kc) {
            v16bf pA;
            const float* prow = &Pl[((m * 16) + ln16) * PL_ROW + kc * 32 + d0];
#pragma unroll
            for (int e = 0; e < 8; ++e) {
                pA[e]     = (bf16_t)prow[e];
                pA[8 + e] = (bf16_t)prow[16 + e];
            }
            const int koff = kc * 32 + lhi * 8;
#pragma unroll
            for (int ct = 0; ct < 8; ++ct) {
                const int c = cbase + ct * 16 + ln16;
                const bf16_t* vb = &Vt[((cur * 256) + c) * VT_ROW + koff];
                union { v16bf f; uint4 u[2]; } r;
                r.u[0] = *(const uint4*)(vb);
                r.u[1] = *(const uint4*)(vb + 16);
                accO[ct] = __builtin_amdgcn_wmma_f32_16x16x32_bf16(
                    false, pA, false, r.f, (short)0, accO[ct], false, false);
            }
        }

        wait_async0();        // next tile landed (this wave's async ops)
        __syncthreads();      // ... and everyone else's
        cur ^= 1;
    }

    // ---- epilogue: out = x + gamma * O / l ----
    float* outp = out + (size_t)vsel * BATCH * CCH * NPIX;
    const size_t xb = (size_t)b * CCH * NPIX;
#pragma unroll
    for (int ct = 0; ct < 8; ++ct) {
        const int c = cbase + ct * 16 + ln16;
#pragma unroll
        for (int r = 0; r < 8; ++r) {
            const int i = i0 + r + 8 * lhi;
            const size_t idx = xb + (size_t)c * NPIX + i;
            outp[idx] = xres[idx] + gamma * (accO[ct][r] / lrow[r]);
        }
    }
}

// ---------------------------------------------------------------------------
extern "C" void kernel_launch(void* const* d_in, const int* in_sizes, int n_in,
                              void* d_out, int out_size, void* d_ws, size_t ws_size,
                              hipStream_t stream) {
    const float* x1  = (const float*)d_in[0];
    const float* x2  = (const float*)d_in[1];
    const float* chg = (const float*)d_in[2];
    const float* Wq  = (const float*)d_in[3];
    const float* bq  = (const float*)d_in[4];
    const float* Wk  = (const float*)d_in[5];
    const float* bk  = (const float*)d_in[6];
    const float* Wv1 = (const float*)d_in[7];
    const float* bv1 = (const float*)d_in[8];
    const float* Wv2 = (const float*)d_in[9];
    const float* bv2 = (const float*)d_in[10];
    const float* g1  = (const float*)d_in[11];
    const float* g2  = (const float*)d_in[12];

    char* ws = (char*)d_ws;
    bf16_t* qT = (bf16_t*)(ws);
    bf16_t* kT = (bf16_t*)(ws + (size_t)1 * 1048576);
    bf16_t* v1 = (bf16_t*)(ws + (size_t)2 * 1048576);
    bf16_t* v2 = (bf16_t*)(ws + (size_t)10 * 1048576);

    const dim3 blk(256);
    const dim3 gq(BATCH * (NPIX / 64), 1);
    const dim3 gv(BATCH * (NPIX / 64), 8);

    proj_bf16_kernel<<<gq, blk, 0, stream>>>(chg, Wq, bq, qT, DQK, 1);
    proj_bf16_kernel<<<gq, blk, 0, stream>>>(chg, Wk, bk, kT, DQK, 1);
    proj_bf16_kernel<<<gv, blk, 0, stream>>>(x1, Wv1, bv1, v1, CCH, 0);
    proj_bf16_kernel<<<gv, blk, 0, stream>>>(x2, Wv2, bv2, v2, CCH, 0);

    const dim3 ga(BATCH * (NPIX / 64), 2);
    attn_flash_kernel<<<ga, blk, SMEM_BYTES, stream>>>(qT, kT, v1, v2, x1, x2,
                                                       g1, g2, (float*)d_out);
}
